// FEARecLayer_19731079758235
// MI455X (gfx1250) — matrix operand order repaired and checked by reading, not compiled, not run on hardware
//
#include <hip/hip_runtime.h>
#include <hip/hip_bf16.h>

// ---------------------------------------------------------------------------
// FEARec layer on MI455X (gfx1250, wave32, WMMA).
// FFTs removed analytically: band-limit filter == circulant GEMM (G symmetric);
// corr mean harvested from attention score diagonals. All heavy math through
// v_wmma_f32_16x16x32_bf16 (fp32 accumulate). All WMMA fragments are loaded as
// contiguous 16B vectors; dense GEMMs stream operands straight from L2 (192MB)
// with no LDS staging. The attention kernel stages Q/K tiles into LDS with the
// CDNA5 async DMA path (global_load_async_to_lds_b128 + s_wait_asynccnt).
// ---------------------------------------------------------------------------

#define B_   32
#define L_   512
#define HID_ 512
#define H_   8
#define E_   64
#define TOPK_ 62

typedef __attribute__((ext_vector_type(16))) __bf16 v16bf;
typedef __attribute__((ext_vector_type(8)))  float  v8f;

union FragBF {
    v16bf v;
    uint4 q[2];
    unsigned short u[16];
};

__device__ __forceinline__ unsigned short f2bf(float f) {
    union { float f; unsigned u; } x; x.f = f;
    unsigned r = x.u + 0x7FFFu + ((x.u >> 16) & 1u);   // RNE
    return (unsigned short)(r >> 16);
}

// async 16B global -> LDS copy (per-lane). For LDS pointers the flat address
// low 32 bits are the LDS byte offset (ISA 10.2).
__device__ __forceinline__ void async_copy_b128(void* lds_dst, const void* gsrc) {
    unsigned lds_off = (unsigned)(unsigned long long)lds_dst;
    asm volatile("global_load_async_to_lds_b128 %0, %1, off"
                 :: "v"(lds_off), "v"(gsrc) : "memory");
}
__device__ __forceinline__ void async_wait0() {
    asm volatile("s_wait_asynccnt 0x0" ::: "memory");
}

// ------------------------------ prep kernels -------------------------------

__global__ void gvec_kernel(float* __restrict__ g) {
    int d = threadIdx.x;                 // 512 threads
    float s = 0.f;
    for (int f = 102; f <= 255; ++f) {   // band [102,257): doubled + Nyquist
        int p = (f * d) & 511;           // exact range reduction
        s += cosf((float)p * 0.012271846303085129f);  // 2*pi/512
    }
    float nyq = (d & 1) ? -1.f : 1.f;
    g[d] = (2.f * s + nyq) * (1.0f / 512.0f);
}

__global__ void gmat_kernel(const float* __restrict__ g, unsigned short* __restrict__ G) {
    int idx = blockIdx.x * 256 + threadIdx.x;    // 512*512
    int n = idx >> 9, m = idx & 511;
    G[idx] = f2bf(g[(n - m) & 511]);             // circulant, symmetric
}

// plain bf16 cast of the four weight matrices: W[j][c] is already the
// N-major ([n][k]) layout the B-fragments want for X @ W^T.
__global__ void wb_kernel(const float* __restrict__ Wq, const float* __restrict__ Wk,
                          const float* __restrict__ Wv, const float* __restrict__ Wd,
                          unsigned short* __restrict__ WB) {
    long gid = (long)blockIdx.x * 256 + threadIdx.x;   // 4 * 512*512
    int wsel = (int)(gid >> 18);
    long rem = gid & 262143;
    const float* W = (wsel == 0) ? Wq : (wsel == 1) ? Wk : (wsel == 2) ? Wv : Wd;
    WB[gid] = f2bf(W[rem]);
}

__global__ void cast_f32_bf16_kernel(const float* __restrict__ src,
                                     unsigned short* __restrict__ dst, long n) {
    long i = (long)blockIdx.x * 256 + threadIdx.x;
    if (i < n) dst[i] = f2bf(src[i]);
}

// per-batch transpose-cast: XT[b][c][m] = X[b][m][c]  (N-major operand for the
// circulant filter GEMM)
__global__ void xt_kernel(const float* __restrict__ X, unsigned short* __restrict__ XT) {
    long gid = (long)blockIdx.x * 256 + threadIdx.x;   // B*L*HID
    int b = (int)(gid >> 18);
    long rem = gid & 262143;
    int c = (int)(rem >> 9), m = (int)(rem & 511);
    XT[gid] = f2bf(X[(long)b * 262144 + (long)m * 512 + c]);
}

// ------------------------------ GEMM (bf16 WMMA) ---------------------------
// C[M,N] = A[M,K] * B[K,N] (+bias).  A given row-major [row][k]; B given
// N-major [n][k] (so both fragment types are contiguous 16B lane loads).
// Block tile 128x128; 8 waves = 4 row-strips x 2 col-strips; 2x4 subtiles.
// No LDS: L2 (192MB) carries all reuse for these 512-wide operands.

__global__ __launch_bounds__(256)
void gemm_bf16_wmma(const unsigned short* __restrict__ A,
                    const unsigned short* __restrict__ BN,
                    void* __restrict__ Cv,
                    int M, int N, int K,
                    long sA, long sB, long sC,
                    const float* __restrict__ bias,
                    int out_bf16, int out_trans, int ldct) {
    const int tid  = threadIdx.x;
    const int lane = tid & 31;
    const int wid  = tid >> 5;
    const int wrow = wid & 3;      // 32-row strip
    const int wcol = wid >> 2;     // 64-col strip
    const int kh   = lane >> 4;
    const int ml   = lane & 15;

    const unsigned short* Ab = A  + (long)blockIdx.z * sA;
    const unsigned short* Bb = BN + (long)blockIdx.z * sB;
    const int row0 = blockIdx.y * 128;
    const int col0 = blockIdx.x * 128;

    const unsigned short* arow[2];
    arow[0] = Ab + (long)(row0 + wrow * 32 + ml) * K;
    arow[1] = Ab + (long)(row0 + wrow * 32 + 16 + ml) * K;
    const unsigned short* brow[4];
    #pragma unroll
    for (int c = 0; c < 4; ++c)
        brow[c] = Bb + (long)(col0 + wcol * 64 + c * 16 + ml) * K;

    v8f acc[2][4];
    for (int s = 0; s < 2; ++s)
        for (int c = 0; c < 4; ++c)
            for (int r = 0; r < 8; ++r) acc[s][c][r] = 0.0f;

    for (int kk = 0; kk < K; kk += 32) {
        if (kk + 32 < K) {                               // L2 prefetch of next slab
            __builtin_prefetch(arow[0] + kk + 32, 0, 1);
            __builtin_prefetch(brow[0] + kk + 32, 0, 1);
        }
        FragBF a[2], b[4];
        #pragma unroll
        for (int s = 0; s < 2; ++s) {
            a[s].q[0] = *(const uint4*)(arow[s] + kk + 8 * kh);        // k = 8kh..+7
            a[s].q[1] = *(const uint4*)(arow[s] + kk + 16 + 8 * kh);   // k = 16+8kh..+7
        }
        #pragma unroll
        for (int c = 0; c < 4; ++c) {
            b[c].q[0] = *(const uint4*)(brow[c] + kk + 8 * kh);        // k = 8kh..+7
            b[c].q[1] = *(const uint4*)(brow[c] + kk + 16 + 8 * kh);   // k = 16+8kh..+7
        }
        #pragma unroll
        for (int s = 0; s < 2; ++s)
            #pragma unroll
            for (int c = 0; c < 4; ++c)
                acc[s][c] = __builtin_amdgcn_wmma_f32_16x16x32_bf16(
                    false, a[s].v, false, b[c].v, (short)0, acc[s][c], false, false);
    }

    for (int s = 0; s < 2; ++s)
        for (int c = 0; c < 4; ++c)
            for (int r = 0; r < 8; ++r) {
                int grow = row0 + wrow * 32 + s * 16 + r + 8 * kh;
                int gcol = col0 + wcol * 64 + c * 16 + ml;
                float v = acc[s][c][r];
                if (bias) v += bias[gcol];
                long idx = (long)blockIdx.z * sC +
                           (out_trans ? ((long)gcol * ldct + grow)
                                      : ((long)grow * N + gcol));
                if (out_bf16) ((unsigned short*)Cv)[idx] = f2bf(v);
                else          ((float*)Cv)[idx] = v;
            }
}

// --------------------- fused spatial attention + corr-mean -----------------
// block = (l-tile of 64, h, b); flash-style online softmax. Also accumulates
// mean_value[b,tau] = (1/512) * sum over heads of raw score diagonals.
// Q/K tiles staged into LDS via CDNA5 async DMA; V consumed transposed from
// global (VST[b][j][l]) so P@V B-fragments are contiguous 16B loads.

__global__ __launch_bounds__(256)
void attn_kernel(const unsigned short* __restrict__ Qs,
                 const unsigned short* __restrict__ Ks,
                 const unsigned short* __restrict__ VST,
                 const float* __restrict__ mask,
                 float* __restrict__ Ctx,
                 float* __restrict__ mv) {
    const int b = blockIdx.z, h = blockIdx.y, l0 = blockIdx.x * 64;
    __shared__ __align__(16) unsigned short Qt[64 * 72];
    __shared__ __align__(16) unsigned short Kt[64 * 72];
    __shared__ __align__(16) unsigned short Pt[64 * 72];
    __shared__ float Sb[64 * 68];
    __shared__ float mvloc[512];
    __shared__ float rowmax[64], rowsum[64], alphas[64];

    const int tid  = threadIdx.x;
    const int lane = tid & 31;
    const int wid  = tid >> 5;
    const int wrow = wid & 3;      // 16 q-rows per wave
    const int wcol = wid >> 2;     // 32 cols per wave
    const int kh   = lane >> 4;
    const int ml   = lane & 15;
    const int qrow = wrow * 16 + ml;

    mvloc[tid] = 0.f; mvloc[tid + 256] = 0.f;
    if (tid < 64) { rowmax[tid] = -1e30f; rowsum[tid] = 0.f; }

    // async-stage Q tile (64 x 64 bf16); completion enforced before first use
    #pragma unroll
    for (int t = 0; t < 2; ++t) {
        int ch = tid + t * 256;
        int r = ch >> 3, q = ch & 7;
        async_copy_b128(&Qt[r * 72 + q * 8],
                        Qs + ((long)b * L_ + l0 + r) * HID_ + h * E_ + q * 8);
    }

    const unsigned short* vbase = VST + ((long)b * HID_ + h * E_) * L_;

    v8f o_acc[2];
    for (int c = 0; c < 2; ++c)
        for (int r = 0; r < 8; ++r) o_acc[c][r] = 0.0f;

    for (int mb = 0; mb < 8; ++mb) {
        const int m0 = mb * 64;
        __syncthreads();               // prior readers of Kt/Pt done
        #pragma unroll
        for (int t = 0; t < 2; ++t) {
            int ch = tid + t * 256;
            int r = ch >> 3, q = ch & 7;
            async_copy_b128(&Kt[r * 72 + q * 8],
                            Ks + ((long)b * L_ + m0 + r) * HID_ + h * E_ + q * 8);
        }
        async_wait0();                 // my async DMAs landed in LDS
        __syncthreads();               // everyone's tiles visible

        // S tile = Q * K^T  (two K=32 steps over e); all fragments 16B loads
        v8f s_acc[2];
        for (int c = 0; c < 2; ++c)
            for (int r = 0; r < 8; ++r) s_acc[c][r] = 0.0f;
        #pragma unroll
        for (int e0 = 0; e0 < 64; e0 += 32) {
            FragBF a, bf[2];
            a.q[0] = *(const uint4*)(&Qt[qrow * 72 + e0 + 8 * kh]);
            a.q[1] = *(const uint4*)(&Qt[qrow * 72 + e0 + 16 + 8 * kh]);
            #pragma unroll
            for (int c = 0; c < 2; ++c) {
                int mcol = wcol * 32 + c * 16 + ml;   // B n-index = key position
                bf[c].q[0] = *(const uint4*)(&Kt[mcol * 72 + e0 + 8 * kh]);
                bf[c].q[1] = *(const uint4*)(&Kt[mcol * 72 + e0 + 16 + 8 * kh]);
            }
            #pragma unroll
            for (int c = 0; c < 2; ++c)
                s_acc[c] = __builtin_amdgcn_wmma_f32_16x16x32_bf16(
                    false, a.v, false, bf[c].v, (short)0, s_acc[c], false, false);
        }

        // corr-mean diagonals + spill scaled/masked scores to LDS
        for (int c = 0; c < 2; ++c)
            for (int r = 0; r < 8; ++r) {
                int ll = wrow * 16 + r + 8 * kh;
                int mm = wcol * 32 + c * 16 + ml;
                float v = s_acc[c][r];
                atomicAdd(&mvloc[(l0 + ll - (m0 + mm)) & 511], v);
                Sb[ll * 68 + mm] = v * 0.125f +
                    mask[((long)b * L_ + (l0 + ll)) * L_ + (m0 + mm)];
            }
        __syncthreads();

        // online softmax update (one thread per q-row); P packed as u32 pairs
        if (tid < 64) {
            float mo = rowmax[tid], mn = mo;
            for (int m = 0; m < 64; ++m) mn = fmaxf(mn, Sb[tid * 68 + m]);
            float al = __expf(mo - mn);
            float s = rowsum[tid] * al;
            unsigned* prow = (unsigned*)&Pt[tid * 72];
            for (int m = 0; m < 32; ++m) {
                float p0 = __expf(Sb[tid * 68 + 2 * m]     - mn);
                float p1 = __expf(Sb[tid * 68 + 2 * m + 1] - mn);
                s += p0 + p1;
                prow[m] = (unsigned)f2bf(p0) | ((unsigned)f2bf(p1) << 16);
            }
            rowmax[tid] = mn; rowsum[tid] = s; alphas[tid] = al;
        }
        __syncthreads();

        // rescale running O, then O += P * V (V fragments straight from global)
        for (int c = 0; c < 2; ++c)
            for (int r = 0; r < 8; ++r)
                o_acc[c][r] *= alphas[wrow * 16 + r + 8 * kh];
        #pragma unroll
        for (int k0 = 0; k0 < 64; k0 += 32) {
            FragBF a, bf[2];
            a.q[0] = *(const uint4*)(&Pt[qrow * 72 + k0 + 8 * kh]);
            a.q[1] = *(const uint4*)(&Pt[qrow * 72 + k0 + 16 + 8 * kh]);
            #pragma unroll
            for (int c = 0; c < 2; ++c) {
                int ecol = wcol * 32 + c * 16 + ml;   // B n-index = e channel
                const unsigned short* vr = vbase + (long)ecol * L_ + m0 + k0 + 8 * kh;
                bf[c].q[0] = *(const uint4*)(vr);
                bf[c].q[1] = *(const uint4*)(vr + 16);
            }
            #pragma unroll
            for (int c = 0; c < 2; ++c)
                o_acc[c] = __builtin_amdgcn_wmma_f32_16x16x32_bf16(
                    false, a.v, false, bf[c].v, (short)0, o_acc[c], false, false);
        }
    }
    __syncthreads();

    // write 0.5 * softmax(S) V into context (disjoint slice per (b,h,l-tile))
    for (int c = 0; c < 2; ++c)
        for (int r = 0; r < 8; ++r) {
            int ll = wrow * 16 + r + 8 * kh;
            int ee = wcol * 32 + c * 16 + ml;
            float v = o_acc[c][r] / rowsum[ll];
            Ctx[((long)b * L_ + l0 + ll) * HID_ + h * E_ + ee] = 0.5f * v;
        }
    // flush corr-mean contribution (mean over H*E = 512 channels)
    atomicAdd(&mv[b * 512 + tid],       mvloc[tid]       * (1.0f / 512.0f));
    atomicAdd(&mv[b * 512 + tid + 256], mvloc[tid + 256] * (1.0f / 512.0f));
}

// ------------------------------ top-k + softmax ----------------------------

__global__ __launch_bounds__(256)
void topk_kernel(const float* __restrict__ mv, int* __restrict__ delays,
                 float* __restrict__ wsm) {
    const int b = blockIdx.x, tid = threadIdx.x;
    __shared__ float vals[512];
    __shared__ float rv[256];
    __shared__ int   ri[256];
    __shared__ float wv[TOPK_];
    vals[tid] = mv[b * 512 + tid];
    vals[tid + 256] = mv[b * 512 + tid + 256];
    __syncthreads();
    for (int k = 0; k < TOPK_; ++k) {
        float v0 = vals[tid], v1 = vals[tid + 256];
        if (v1 > v0) { rv[tid] = v1; ri[tid] = tid + 256; }
        else         { rv[tid] = v0; ri[tid] = tid; }
        __syncthreads();
        for (int s = 128; s > 0; s >>= 1) {
            if (tid < s) {
                if (rv[tid + s] > rv[tid] ||
                    (rv[tid + s] == rv[tid] && ri[tid + s] < ri[tid])) {
                    rv[tid] = rv[tid + s]; ri[tid] = ri[tid + s];
                }
            }
            __syncthreads();
        }
        if (tid == 0) {
            delays[b * TOPK_ + k] = ri[0];
            wv[k] = rv[0];
            vals[ri[0]] = -3.4e38f;
        }
        __syncthreads();
    }
    if (tid == 0) {
        float mx = wv[0], s = 0.f;                 // descending order
        for (int k = 0; k < TOPK_; ++k) { float e = __expf(wv[k] - mx); wv[k] = e; s += e; }
        float inv = 1.0f / s;
        for (int k = 0; k < TOPK_; ++k) wsm[b * TOPK_ + k] = wv[k] * inv;
    }
}

// --------------------------- time-delay aggregation ------------------------

__global__ __launch_bounds__(256)
void delay_agg_kernel(const float* __restrict__ Vf, const int* __restrict__ delays,
                      const float* __restrict__ wsm, float* __restrict__ Ctx) {
    const int b = blockIdx.y, n = blockIdx.x, tid = threadIdx.x;
    __shared__ int   dl[TOPK_];
    __shared__ float wl[TOPK_];
    if (tid < TOPK_) { dl[tid] = delays[b * TOPK_ + tid]; wl[tid] = wsm[b * TOPK_ + tid]; }
    __syncthreads();
    float a0 = 0.f, a1 = 0.f;
    for (int k = 0; k < TOPK_; ++k) {
        int row = (n + dl[k]) & 511;
        const float* src = Vf + ((long)b * L_ + row) * HID_;
        a0 += wl[k] * src[tid];
        a1 += wl[k] * src[tid + 256];
    }
    long base = ((long)b * L_ + n) * HID_;
    Ctx[base + tid]       += 0.5f * a0;
    Ctx[base + tid + 256] += 0.5f * a1;
}

// ----------------------- residual + layernorm epilogue ---------------------

__global__ __launch_bounds__(256)
void epilogue_kernel(const float* __restrict__ Hpre, const float* __restrict__ X,
                     const float* __restrict__ bd, const float* __restrict__ lnw,
                     const float* __restrict__ lnb, float* __restrict__ out) {
    const int b = blockIdx.y, n = blockIdx.x, tid = threadIdx.x;
    __shared__ float red[256];
    long base = ((long)b * L_ + n) * HID_;
    float h0 = Hpre[base + tid]       + bd[tid]       + X[base + tid];
    float h1 = Hpre[base + tid + 256] + bd[tid + 256] + X[base + tid + 256];
    red[tid] = h0 + h1;
    __syncthreads();
    for (int s = 128; s > 0; s >>= 1) { if (tid < s) red[tid] += red[tid + s]; __syncthreads(); }
    float u = red[0] * (1.0f / 512.0f);
    __syncthreads();
    red[tid] = (h0 - u) * (h0 - u) + (h1 - u) * (h1 - u);
    __syncthreads();
    for (int s = 128; s > 0; s >>= 1) { if (tid < s) red[tid] += red[tid + s]; __syncthreads(); }
    float inv = rsqrtf(red[0] * (1.0f / 512.0f) + 1e-12f);
    out[base + tid]       = lnw[tid]       * (h0 - u) * inv + lnb[tid];
    out[base + tid + 256] = lnw[tid + 256] * (h1 - u) * inv + lnb[tid + 256];
}

// ------------------------------- launcher ----------------------------------

extern "C" void kernel_launch(void* const* d_in, const int* in_sizes, int n_in,
                              void* d_out, int out_size, void* d_ws, size_t ws_size,
                              hipStream_t stream) {
    const float* X   = (const float*)d_in[0];
    const float* msk = (const float*)d_in[1];
    const float* Wq  = (const float*)d_in[2];
    const float* Wk  = (const float*)d_in[4];
    const float* Wv  = (const float*)d_in[6];
    const float* bv  = (const float*)d_in[7];
    const float* Wd  = (const float*)d_in[8];
    const float* bd  = (const float*)d_in[9];
    const float* lnw = (const float*)d_in[10];
    const float* lnb = (const float*)d_in[11];

    char* ws = (char*)d_ws;
    const size_t MB = 1u << 20;
    unsigned short* XB   = (unsigned short*)(ws + 0);              // 16 MB bf16 X
    unsigned short* WB   = (unsigned short*)(ws + 16 * MB);        // 2 MB  bf16 weights
    float*          GVEC = (float*)(ws + 18 * MB);                 // 2 KB
    unsigned short* GMAT = (unsigned short*)(ws + 18 * MB + 8192); // 0.5 MB circulant
    unsigned short* XF   = (unsigned short*)(ws + 19 * MB);        // 16 MB filtered X
    unsigned short* QS   = (unsigned short*)(ws + 35 * MB);        // 16 MB
    unsigned short* KS   = (unsigned short*)(ws + 51 * MB);        // 16 MB
    unsigned short* VST  = (unsigned short*)(ws + 67 * MB);        // 16 MB V filtered, [b][j][l]
    float*          VF   = (float*)(ws + 83 * MB);                 // 32 MB unfiltered V (fp32)
    float*          MV   = (float*)(ws + 115 * MB);                // 64 KB mean_value
    int*            DL   = (int*)(ws + 115 * MB + 65536);
    float*          WSM  = (float*)(ws + 115 * MB + 131072);
    float*          CTX  = (float*)(ws + 116 * MB);                // 32 MB combined context
    unsigned short* XT   = (unsigned short*)(ws + 116 * MB);       // 16 MB X^T (dead before CTX)
    unsigned short* CTXB = XF;                                     // reuse: bf16 context
    float*          HPRE = (float*)QS;                             // reuse QS+KS: 32 MB pre-LN

    const long NELEM = (long)B_ * L_ * HID_;                       // 8388608

    hipMemsetAsync(MV, 0, B_ * 512 * sizeof(float), stream);

    gvec_kernel<<<1, 512, 0, stream>>>(GVEC);
    gmat_kernel<<<1024, 256, 0, stream>>>(GVEC, GMAT);
    wb_kernel<<<4096, 256, 0, stream>>>(Wq, Wk, Wv, Wd, WB);
    cast_f32_bf16_kernel<<<(int)(NELEM / 256), 256, 0, stream>>>(X, XB, NELEM);
    xt_kernel<<<(int)(NELEM / 256), 256, 0, stream>>>(X, XT);

    // Xf[b] = G @ X[b]  (circulant filter; G symmetric, X^T is the N-major operand)
    gemm_bf16_wmma<<<dim3(4, 4, 32), 256, 0, stream>>>(
        GMAT, XT, XF, 512, 512, 512, 0, 262144, 262144, nullptr, 1, 0, 0);
    // Qs/Ks = Xf @ W^T  (bias filtered out: band excludes DC)
    gemm_bf16_wmma<<<dim3(4, 128, 1), 256, 0, stream>>>(
        XF, WB + 0 * 262144, QS, 16384, 512, 512, 0, 0, 0, nullptr, 1, 0, 0);
    gemm_bf16_wmma<<<dim3(4, 128, 1), 256, 0, stream>>>(
        XF, WB + 1 * 262144, KS, 16384, 512, 512, 0, 0, 0, nullptr, 1, 0, 0);
    // Vs = Xf @ Wv^T, written transposed per batch: VST[b][j][l]
    gemm_bf16_wmma<<<dim3(4, 4, 32), 256, 0, stream>>>(
        XF, WB + 2 * 262144, VST, 512, 512, 512, 262144, 0, 262144, nullptr, 1, 1, 512);
    // V_full = X @ Wv^T + bv  (fp32, feeds time-delay aggregation)
    gemm_bf16_wmma<<<dim3(4, 128, 1), 256, 0, stream>>>(
        XB, WB + 2 * 262144, VF, 16384, 512, 512, 0, 0, 0, bv, 0, 0, 0);

    // spatial attention + corr-mean diagonals
    attn_kernel<<<dim3(8, 8, 32), 256, 0, stream>>>(QS, KS, VST, msk, CTX, MV);
    // top-62 delays + softmax weights
    topk_kernel<<<32, 256, 0, stream>>>(MV, DL, WSM);
    // context += 0.5 * time_delay_agg(V_full)
    delay_agg_kernel<<<dim3(512, 32), 256, 0, stream>>>(VF, DL, WSM, CTX);

    // output projection + residual + layernorm
    cast_f32_bf16_kernel<<<(int)(NELEM / 256), 256, 0, stream>>>(CTX, CTXB, NELEM);
    gemm_bf16_wmma<<<dim3(4, 128, 1), 256, 0, stream>>>(
        CTXB, WB + 3 * 262144, HPRE, 16384, 512, 512, 0, 0, 0, nullptr, 0, 0, 0);
    epilogue_kernel<<<dim3(512, 32), 256, 0, stream>>>(HPRE, X, bd, lnw, lnb, (float*)d_out);
}
